// Encoder_14663018348865
// MI455X (gfx1250) — matrix operand order repaired and checked by reading, not compiled, not run on hardware
//
#include <hip/hip_runtime.h>
#include <math.h>

typedef __bf16 bf16;
typedef __attribute__((ext_vector_type(16))) __bf16 bf16x16;
typedef __attribute__((ext_vector_type(8)))  __bf16 bf16x8;
typedef __attribute__((ext_vector_type(8)))  float  f32x8;

#define BB 8
#define SS 1024
#define FF 8
#define AA 8
#define EE 64
#define MM 512
#define HH 8
#define HT 128
#define DD 192

#define BPAD 40   // LDS B-tile row stride in bf16 elements (80B = 20 banks, conflict-free)

__device__ __forceinline__ bf16 f2bf(float f) {
  union { float f; unsigned u; } v; v.f = f;
  unsigned r = v.u + 0x7FFFu + ((v.u >> 16) & 1u);
  union { unsigned short s; bf16 b; } o; o.s = (unsigned short)(r >> 16);
  return o.b;
}

__device__ __forceinline__ bf16x16 cat8(bf16x8 lo, bf16x8 hi) {
  bf16x16 r;
#pragma unroll
  for (int i = 0; i < 8; ++i) { r[i] = lo[i]; r[i + 8] = hi[i]; }
  return r;
}

__device__ __forceinline__ f32x8 zero8() {
  f32x8 z = {0.f,0.f,0.f,0.f,0.f,0.f,0.f,0.f};
  return z;
}

__device__ __forceinline__ f32x8 wmma_bf16(bf16x16 a, bf16x16 b, f32x8 c) {
  // D = A(16x32 bf16) * B(32x16 bf16) + C(16x16 f32)
  return __builtin_amdgcn_wmma_f32_16x16x32_bf16(false, a, false, b, (short)0, c, false, false);
}

// Async DMA: global -> LDS, 16 bytes per lane, tracked by ASYNCcnt.
__device__ __forceinline__ void async_g2l_b128(unsigned lds_off, const bf16* gaddr) {
  asm volatile("global_load_async_to_lds_b128 %0, %1, off"
               :: "v"(lds_off), "v"(gaddr) : "memory");
}
__device__ __forceinline__ void wait_async_le2() {
  asm volatile("s_wait_asynccnt 0x2" ::: "memory");
}
__device__ __forceinline__ void wait_async_0() {
  asm volatile("s_wait_asynccnt 0x0" ::: "memory");
}

// ---------------------------------------------------------------------------
// Stage 1+2: VarEmbed (per-feature MLPs + mask select) and feature attention.
// One block of 128 threads per (b,s) token. Writes x = r.reshape(B,S,M) (f32).
// ---------------------------------------------------------------------------
__global__ void var_token_kernel(
    const float* __restrict__ inputs, const float* __restrict__ aux,
    const unsigned char* __restrict__ mask,
    const float* __restrict__ pOw, const float* __restrict__ pOb,
    const float* __restrict__ pMw, const float* __restrict__ pMb,
    const float* __restrict__ eOw, const float* __restrict__ eOb,
    const float* __restrict__ eMw, const float* __restrict__ eMb,
    const float* __restrict__ lng, const float* __restrict__ lnb,
    const float* __restrict__ qw, const float* __restrict__ qb2,
    const float* __restrict__ kw, const float* __restrict__ kb2,
    const float* __restrict__ vw, const float* __restrict__ vb2,
    const float* __restrict__ ow, const float* __restrict__ ob2,
    float* __restrict__ xout) {
  __shared__ float auxs[AA], xv[FF];
  __shared__ float hidO[128], hidM[128];
  __shared__ float rbuf[FF][EE], ybuf[FF][EE];
  __shared__ float qs[FF][EE], ks[FF][EE], vs[FF][EE];
  __shared__ float att[HH][FF][FF];
  __shared__ float obuf[FF][EE];
  __shared__ float stat[FF][2];
  int tid = threadIdx.x;
  int b = blockIdx.x >> 10, s = blockIdx.x & 1023;
  if (tid < AA) auxs[tid] = aux[((size_t)b * AA + tid) * SS + s];
  if (tid < FF) xv[tid]   = inputs[((size_t)b * FF + tid) * SS + s];
  __syncthreads();
  for (int f = 0; f < FF; ++f) {
    {
      float aO = pOb[f * 128 + tid];
      aO += xv[f] * pOw[((size_t)f * 9 + 0) * 128 + tid];
#pragma unroll
      for (int a = 0; a < AA; ++a) aO += auxs[a] * pOw[((size_t)f * 9 + 1 + a) * 128 + tid];
      hidO[tid] = fmaxf(aO, 0.0f);
      float aM = pMb[f * 128 + tid];
#pragma unroll
      for (int a = 0; a < AA; ++a) aM += auxs[a] * pMw[((size_t)f * AA + a) * 128 + tid];
      hidM[tid] = fmaxf(aM, 0.0f);
    }
    __syncthreads();
    if (tid < EE) {
      float aO = eOb[f * EE + tid], aM = eMb[f * EE + tid];
      for (int t = 0; t < 128; ++t) {
        aO += hidO[t] * eOw[((size_t)f * 128 + t) * EE + tid];
        aM += hidM[t] * eMw[((size_t)f * 128 + t) * EE + tid];
      }
      bool m = mask[((size_t)b * FF + f) * SS + s] != 0;
      rbuf[f][tid] = m ? aO : aM;
    }
    __syncthreads();
  }
  if (tid < FF) {
    float mu = 0.f;
    for (int e = 0; e < EE; ++e) mu += rbuf[tid][e];
    mu *= (1.0f / EE);
    float vv = 0.f;
    for (int e = 0; e < EE; ++e) { float d = rbuf[tid][e] - mu; vv += d * d; }
    stat[tid][0] = mu;
    stat[tid][1] = rsqrtf(vv * (1.0f / EE) + 1e-5f);
  }
  __syncthreads();
  for (int idx = tid; idx < FF * EE; idx += 128) {
    int f = idx >> 6, e = idx & 63;
    ybuf[f][e] = (rbuf[f][e] - stat[f][0]) * stat[f][1] * lng[e] + lnb[e];
  }
  __syncthreads();
  for (int idx = tid; idx < FF * EE; idx += 128) {
    int f = idx >> 6, e = idx & 63;
    float aq = qb2[e], ak = kb2[e], av = vb2[e];
    for (int i = 0; i < EE; ++i) {
      float yv = ybuf[f][i];
      aq += yv * qw[i * EE + e];
      ak += yv * kw[i * EE + e];
      av += yv * vw[i * EE + e];
    }
    qs[f][e] = aq; ks[f][e] = ak; vs[f][e] = av;
  }
  __syncthreads();
  for (int idx = tid; idx < HH * FF * FF; idx += 128) {
    int h = idx >> 6, f = (idx >> 3) & 7, g = idx & 7;
    float sum = 0.f;
#pragma unroll
    for (int d = 0; d < 8; ++d) sum += qs[f][h * 8 + d] * ks[g][h * 8 + d];
    att[h][f][g] = sum * 0.35355339059f; // 1/sqrt(8)
  }
  __syncthreads();
  if (tid < HH * FF) {
    int h = tid >> 3, f = tid & 7;
    float mx = -3e38f;
#pragma unroll
    for (int g = 0; g < FF; ++g) mx = fmaxf(mx, att[h][f][g]);
    float sum = 0.f;
#pragma unroll
    for (int g = 0; g < FF; ++g) { float e = __expf(att[h][f][g] - mx); att[h][f][g] = e; sum += e; }
    float inv = 1.0f / sum;
#pragma unroll
    for (int g = 0; g < FF; ++g) att[h][f][g] *= inv;
  }
  __syncthreads();
  for (int idx = tid; idx < FF * EE; idx += 128) {
    int f = idx >> 6, i = idx & 63, h = i >> 3;
    float sum = 0.f;
#pragma unroll
    for (int g = 0; g < FF; ++g) sum += att[h][f][g] * vs[g][i];
    obuf[f][i] = sum;
  }
  __syncthreads();
  for (int idx = tid; idx < FF * EE; idx += 128) {
    int f = idx >> 6, e = idx & 63;
    float acc = ob2[e];
    for (int i = 0; i < EE; ++i) acc += obuf[f][i] * ow[i * EE + e];
    xout[((size_t)b * SS + s) * MM + f * EE + e] = rbuf[f][e] + acc;
  }
}

// ---------------------------------------------------------------------------
// LayerNorm over rows of [8192,512]; writes bf16 normalized activations.
// ---------------------------------------------------------------------------
__global__ void ln_rows_kernel(const float* __restrict__ x, const float* __restrict__ g,
                               const float* __restrict__ bta, bf16* __restrict__ y) {
  __shared__ float red[256];
  int row = blockIdx.x, tid = threadIdx.x;
  const float* xr = x + (size_t)row * MM;
  float s = 0.f;
  for (int i = tid; i < MM; i += 256) s += xr[i];
  red[tid] = s; __syncthreads();
  for (int w = 128; w > 0; w >>= 1) { if (tid < w) red[tid] += red[tid + w]; __syncthreads(); }
  float mu = red[0] * (1.0f / MM);
  __syncthreads();
  float vv = 0.f;
  for (int i = tid; i < MM; i += 256) { float d = xr[i] - mu; vv += d * d; }
  red[tid] = vv; __syncthreads();
  for (int w = 128; w > 0; w >>= 1) { if (tid < w) red[tid] += red[tid + w]; __syncthreads(); }
  float rstd = rsqrtf(red[0] * (1.0f / MM) + 1e-5f);
  for (int i = tid; i < MM; i += 256)
    y[(size_t)row * MM + i] = f2bf((xr[i] - mu) * rstd * g[i] + bta[i]);
}

// ---------------------------------------------------------------------------
// bf16 WMMA GEMM: C[M,N] = (A[M,K](bf16) @ W[K,N] + bias) * alpha + resid
// Wt supplied pre-transposed [N,K] bf16. Block = 4 waves; each wave owns a
// 16-row tile; the 64-column B tile is SHARED and staged in LDS with
// double-buffered GLOBAL_LOAD_ASYNC_TO_LDS_B128 (ASYNCcnt pipelined).
// ---------------------------------------------------------------------------
__global__ void gemm_bf16_kernel(const bf16* __restrict__ A, const bf16* __restrict__ Wt,
                                 const float* __restrict__ bias, const float* __restrict__ resid,
                                 float alpha, float* __restrict__ outF, bf16* __restrict__ outB,
                                 int M, int N, int K) {
  __shared__ bf16 bsh[2][64 * BPAD];  // 2 x 5 KB, padded rows
  int lane = threadIdx.x;
  int tid = threadIdx.y * 32 + lane;
  int half = lane >> 4;
  int lm = lane & 15;
  int m0 = (blockIdx.x * 4 + threadIdx.y) * 16;
  int n0 = blockIdx.y * 64;
  int ka = half * 8;         // A-frag: lanes 0-15 hold K{0..7,16..23}; 16-31 hold K{8..15,24..31}
  int kboff = half * 8;      // B-frag: lanes 0-15 hold K{0..7,16..23}; 16-31 hold K{8..15,24..31}
  int crow = tid >> 2, cseg = tid & 3; // async-copy assignment: rows crow, crow+32; 16B segment cseg

  f32x8 acc[4];
#pragma unroll
  for (int t = 0; t < 4; ++t) acc[t] = zero8();
  const bf16* arow = A + (size_t)(m0 + lm) * K;

  // prologue: issue buffer 0
  {
    const bf16* g0 = Wt + (size_t)(n0 + crow) * K + cseg * 8;
    const bf16* g1 = Wt + (size_t)(n0 + 32 + crow) * K + cseg * 8;
    async_g2l_b128((unsigned)(size_t)&bsh[0][crow * BPAD + cseg * 8], g0);
    async_g2l_b128((unsigned)(size_t)&bsh[0][(crow + 32) * BPAD + cseg * 8], g1);
  }
  int nk = K >> 5;
  for (int ik = 0; ik < nk; ++ik) {
    int kb = ik << 5;
    if (ik + 1 < nk) {
      const bf16* g0 = Wt + (size_t)(n0 + crow) * K + kb + 32 + cseg * 8;
      const bf16* g1 = Wt + (size_t)(n0 + 32 + crow) * K + kb + 32 + cseg * 8;
      int nb = (ik + 1) & 1;
      async_g2l_b128((unsigned)(size_t)&bsh[nb][crow * BPAD + cseg * 8], g0);
      async_g2l_b128((unsigned)(size_t)&bsh[nb][(crow + 32) * BPAD + cseg * 8], g1);
      __builtin_prefetch(arow + kb + 64, 0, 1);  // prefetch next A chunk
      wait_async_le2();                          // previous buffer complete
    } else {
      wait_async_0();
    }
    __syncthreads();
    bf16x16 af = cat8(*(const bf16x8*)(arow + kb + ka),
                      *(const bf16x8*)(arow + kb + ka + 16));
    const bf16* bs = &bsh[ik & 1][0];
#pragma unroll
    for (int t = 0; t < 4; ++t) {
      const bf16* wp = bs + (t * 16 + lm) * BPAD + kboff;
      bf16x16 bfr = cat8(*(const bf16x8*)wp, *(const bf16x8*)(wp + 16));
      acc[t] = wmma_bf16(af, bfr, acc[t]);
    }
    __syncthreads();   // all waves done with this buffer before it is re-filled
  }
#pragma unroll
  for (int t = 0; t < 4; ++t) {
#pragma unroll
    for (int r = 0; r < 8; ++r) {
      int row = m0 + r + half * 8;  // C layout: vgpr r -> rows r / r+8
      int col = n0 + t * 16 + lm;
      float v = acc[t][r];
      if (bias) v += bias[col];
      v *= alpha;
      if (resid) v += resid[(size_t)row * N + col];
      if (outF) outF[(size_t)row * N + col] = v;
      if (outB) outB[(size_t)row * N + col] = f2bf(v);
    }
  }
}

// ---------------------------------------------------------------------------
// Weight transpose + bf16 convert: W[K,N] f32 -> Wt[N,K] bf16
// ---------------------------------------------------------------------------
__global__ void wtrans_kernel(const float* __restrict__ W, bf16* __restrict__ Wt, int K, int N) {
  int i = blockIdx.x * blockDim.x + threadIdx.x;
  if (i >= K * N) return;
  int n = i / K, k = i - n * K;
  Wt[(size_t)n * K + k] = f2bf(W[(size_t)k * N + n]);
}

__global__ void conv_bf16_kernel(const float* __restrict__ src, bf16* __restrict__ dst, int n) {
  int i = blockIdx.x * blockDim.x + threadIdx.x;
  if (i < n) dst[i] = f2bf(src[i]);
}

// v[8192,512] f32 -> vt[B,H,64,S] bf16 (per-head transposed for PV B-fragments)
__global__ void vtrans_kernel(const float* __restrict__ v, bf16* __restrict__ vt) {
  int i = blockIdx.x * blockDim.x + threadIdx.x;
  if (i >= BB * HH * 64 * SS) return;
  int s = i & 1023;
  int rest = i >> 10;
  int d = rest & 63;
  rest >>= 6;
  int h = rest & 7;
  int b = rest >> 3;
  vt[i] = f2bf(v[((size_t)b * SS + s) * MM + h * 64 + d]);
}

// ---------------------------------------------------------------------------
// Temporal attention: one wave per (b,h,16-row block). Full 16x1024 score
// strip staged in dynamic LDS; exact softmax (all 32 lanes, 2 per row);
// PV via WMMA with A-fragments from LDS bf16 P.
// Bias: (j<=i ? 1 : 0) + 2^-(h+1) * j  (additive causal + ALiBi, per ref).
// ---------------------------------------------------------------------------
__global__ void flash_attn_kernel(const bf16* __restrict__ q, const bf16* __restrict__ k,
                                  const bf16* __restrict__ vt, float* __restrict__ o) {
  extern __shared__ char smem[];
  float* sc   = (float*)smem;                            // 16*1024 f32
  bf16* p     = (bf16*)(smem + 16 * 1024 * 4);           // 16*1024 bf16
  float* rinv = (float*)(smem + 16 * 1024 * 4 + 16 * 1024 * 2); // 16 f32
  float* pmax = rinv + 16;                               // 32 f32
  float* psum = pmax + 32;                               // 32 f32
  int lane = threadIdx.x;
  int half = lane >> 4, lm = lane & 15;
  int blk = blockIdx.x;
  int b = blk >> 9;
  int h = (blk >> 6) & 7;
  int i0 = (blk & 63) << 4;
  float slope = exp2f(-(float)(h + 1));
  int ka = half * 8;
  int kboff = half * 8;
  const bf16* qp = q + ((size_t)(b * SS + i0 + lm)) * MM + h * 64;
  bf16x16 aq0 = cat8(*(const bf16x8*)(qp + ka),      *(const bf16x8*)(qp + ka + 16));
  bf16x16 aq1 = cat8(*(const bf16x8*)(qp + 32 + ka), *(const bf16x8*)(qp + 32 + ka + 16));
  // Phase 1: scores
  for (int j0 = 0; j0 < SS; j0 += 16) {
    f32x8 acc = zero8();
    const bf16* kp = k + ((size_t)(b * SS + j0 + lm)) * MM + h * 64 + kboff;
    bf16x16 b0 = cat8(*(const bf16x8*)kp,        *(const bf16x8*)(kp + 16));
    bf16x16 b1 = cat8(*(const bf16x8*)(kp + 32), *(const bf16x8*)(kp + 48));
    acc = wmma_bf16(aq0, b0, acc);
    acc = wmma_bf16(aq1, b1, acc);
#pragma unroll
    for (int r = 0; r < 8; ++r) {
      int lrow = r + half * 8;
      int irow = i0 + lrow;
      int j = j0 + lm;
      sc[lrow * SS + j] = acc[r] + ((j <= irow) ? 1.0f : 0.0f) + slope * (float)j;
    }
  }
  __syncthreads();
  // Phase 2: exact softmax; 2 lanes per row
  {
    int row = lane >> 1;
    int hc = (lane & 1) * 512;
    float mx = -3e38f;
    for (int j = 0; j < 512; ++j) mx = fmaxf(mx, sc[row * SS + hc + j]);
    pmax[lane] = mx;
    __syncthreads();
    mx = fmaxf(pmax[row * 2], pmax[row * 2 + 1]);
    float sum = 0.f;
    for (int j = 0; j < 512; ++j) {
      float e = __expf(sc[row * SS + hc + j] - mx);
      sum += e;
      p[row * SS + hc + j] = f2bf(e);
    }
    psum[lane] = sum;
    __syncthreads();
    if (lane < 16) rinv[lane] = 1.0f / (psum[lane * 2] + psum[lane * 2 + 1]);
  }
  __syncthreads();
  // Phase 3: O = P @ V
  f32x8 oacc[4];
#pragma unroll
  for (int t = 0; t < 4; ++t) oacc[t] = zero8();
  for (int kb = 0; kb < SS; kb += 32) {
    bf16x16 af = cat8(*(const bf16x8*)(p + lm * SS + kb + ka),
                      *(const bf16x8*)(p + lm * SS + kb + ka + 16));
#pragma unroll
    for (int t = 0; t < 4; ++t) {
      const bf16* vp = vt + ((size_t)((b * HH + h) * 64 + t * 16 + lm)) * SS + kb + kboff;
      bf16x16 bfr = cat8(*(const bf16x8*)vp, *(const bf16x8*)(vp + 16));
      oacc[t] = wmma_bf16(af, bfr, oacc[t]);
    }
  }
#pragma unroll
  for (int t = 0; t < 4; ++t) {
#pragma unroll
    for (int r = 0; r < 8; ++r) {
      int lrow = r + half * 8;
      o[((size_t)(b * SS + i0 + lrow)) * MM + h * 64 + t * 16 + lm] = oacc[t][r] * rinv[lrow];
    }
  }
}

// ---------------------------------------------------------------------------
// GRU scan: one block of 1024 threads (8 batches x 128 hidden). W_hh in LDS.
// ---------------------------------------------------------------------------
__global__ void gru_scan_kernel(const float* __restrict__ gx, const float* __restrict__ whh,
                                const float* __restrict__ bhh, float* __restrict__ traj,
                                float* __restrict__ hN) {
  extern __shared__ float sm[];
  float* w = sm;                   // 128*384
  float* h0buf = sm + 128 * 384;   // 8*128
  float* h1buf = h0buf + 8 * 128;  // 8*128
  int tid = threadIdx.x;
  for (int i = tid; i < 128 * 384; i += 1024) w[i] = whh[i];
  int b = tid >> 7, u = tid & 127;
  h0buf[tid] = 0.0f;
  float bb0 = bhh[u], bb1 = bhh[128 + u], bb2 = bhh[256 + u];
  __syncthreads();
  float* hc = h0buf;
  float* hn = h1buf;
  for (int t = 0; t < SS; ++t) {
    const float* hb = hc + b * 128;
    float a0 = bb0, a1 = bb1, a2 = bb2;
    for (int i = 0; i < 128; ++i) {
      float hv = hb[i];
      const float* wr = w + i * 384;
      a0 += hv * wr[u];
      a1 += hv * wr[128 + u];
      a2 += hv * wr[256 + u];
    }
    const float* gxr = gx + ((size_t)b * SS + t) * 384;
    float rr = 1.0f / (1.0f + __expf(-(gxr[u] + a0)));
    float zz = 1.0f / (1.0f + __expf(-(gxr[128 + u] + a1)));
    float nn = tanhf(gxr[256 + u] + rr * a2);
    float hv2 = (1.0f - zz) * nn + zz * hb[u];
    hn[tid] = hv2;
    traj[((size_t)b * SS + t) * HT + u] = hv2;
    __syncthreads();
    float* tmp = hc; hc = hn; hn = tmp;
  }
  hN[tid] = hc[tid];
}

// mean over S of x[b, s, c] -> ragg[b, c]
__global__ void ragg_kernel(const float* __restrict__ x, float* __restrict__ ragg) {
  int i = blockIdx.x * blockDim.x + threadIdx.x;
  if (i >= BB * MM) return;
  int b = i >> 9, c = i & 511;
  float s = 0.f;
  for (int t = 0; t < SS; ++t) s += x[((size_t)b * SS + t) * MM + c];
  ragg[i] = s * (1.0f / SS);
}

// per-(b,f) distribution heads: mu and logsig
__global__ void heads_kernel(const float* __restrict__ ragg, const float* __restrict__ hN,
                             const float* __restrict__ mu_w, const float* __restrict__ mu_b,
                             const float* __restrict__ ls_w, const float* __restrict__ ls_b,
                             float* __restrict__ out) {
  __shared__ float rv[DD];
  int b = blockIdx.x >> 3, f = blockIdx.x & 7;
  int e = threadIdx.x; // 0..191
  if (e < EE) rv[e] = ragg[b * MM + f * EE + e];
  else        rv[e] = hN[b * HT + (e - EE)];
  __syncthreads();
  float am = mu_b[f * DD + e], al = ls_b[f * DD + e];
  const float* mwf = mu_w + (size_t)f * DD * DD;
  const float* lwf = ls_w + (size_t)f * DD * DD;
  for (int d = 0; d < DD; ++d) {
    float rvd = rv[d];
    am += rvd * mwf[d * DD + e];
    al += rvd * lwf[d * DD + e];
  }
  out[b * (FF * DD) + f * DD + e] = am;
  out[BB * FF * DD + b * (FF * DD) + f * DD + e] = al;
}

// ---------------------------------------------------------------------------
extern "C" void kernel_launch(void* const* d_in, const int* in_sizes, int n_in,
                              void* d_out, int out_size, void* d_ws, size_t ws_size,
                              hipStream_t stream) {
  (void)in_sizes; (void)n_in; (void)out_size; (void)ws_size;
  const float* inputs   = (const float*)d_in[0];
  const float* aux      = (const float*)d_in[1];
  const unsigned char* mask = (const unsigned char*)d_in[2];
  const float* pre_O_w  = (const float*)d_in[3];
  const float* pre_O_b  = (const float*)d_in[4];
  const float* pre_M_w  = (const float*)d_in[5];
  const float* pre_M_b  = (const float*)d_in[6];
  const float* emb_O_w  = (const float*)d_in[7];
  const float* emb_O_b  = (const float*)d_in[8];
  const float* emb_M_w  = (const float*)d_in[9];
  const float* emb_M_b  = (const float*)d_in[10];
  const float* ln_v_s   = (const float*)d_in[11];
  const float* ln_v_b   = (const float*)d_in[12];
  const float* vq_w = (const float*)d_in[13]; const float* vq_b = (const float*)d_in[14];
  const float* vk_w = (const float*)d_in[15]; const float* vk_b = (const float*)d_in[16];
  const float* vv_w = (const float*)d_in[17]; const float* vv_b = (const float*)d_in[18];
  const float* vo_w = (const float*)d_in[19]; const float* vo_b = (const float*)d_in[20];
  const float* ln_t_s = (const float*)d_in[21];
  const float* ln_t_b = (const float*)d_in[22];
  const float* tq_w = (const float*)d_in[23]; const float* tq_b = (const float*)d_in[24];
  const float* tk_w = (const float*)d_in[25]; const float* tk_b = (const float*)d_in[26];
  const float* tv_w = (const float*)d_in[27]; const float* tv_b = (const float*)d_in[28];
  const float* to_w = (const float*)d_in[29]; const float* to_b = (const float*)d_in[30];
  const float* gru_wih = (const float*)d_in[31];
  const float* gru_whh = (const float*)d_in[32];
  const float* gru_bih = (const float*)d_in[33];
  const float* gru_bhh = (const float*)d_in[34];
  const float* mu_w = (const float*)d_in[35]; const float* mu_b = (const float*)d_in[36];
  const float* ls_w = (const float*)d_in[37]; const float* ls_b = (const float*)d_in[38];
  float* out = (float*)d_out;

  char* ws = (char*)d_ws;
  const size_t MB16 = (size_t)BB * SS * MM * 4;   // 16 MB
  const size_t MB8  = (size_t)BB * SS * MM * 2;   // 8 MB
  size_t off = 0;
  float* x    = (float*)(ws + off); off += MB16;
  float* x2   = (float*)(ws + off); off += MB16;
  float* vbuf = (float*)(ws + off); off += MB16;
  float* obuf = (float*)(ws + off); off += MB16;
  float* gx   = (float*)(ws + off); off += (size_t)BB * SS * 384 * 4;
  bf16* y_bf  = (bf16*)(ws + off); off += MB8;
  bf16* q_bf  = (bf16*)(ws + off); off += MB8;
  bf16* k_bf  = (bf16*)(ws + off); off += MB8;
  bf16* x2_bf = (bf16*)(ws + off); off += MB8;
  bf16* o_bf  = (bf16*)(ws + off); off += MB8;
  bf16* vt_bf = (bf16*)(ws + off); off += MB8;
  bf16* tqT = (bf16*)(ws + off); off += (size_t)MM * MM * 2;
  bf16* tkT = (bf16*)(ws + off); off += (size_t)MM * MM * 2;
  bf16* tvT = (bf16*)(ws + off); off += (size_t)MM * MM * 2;
  bf16* toT = (bf16*)(ws + off); off += (size_t)MM * MM * 2;
  bf16* wihT = (bf16*)(ws + off); off += (size_t)384 * MM * 2;
  float* ragg = (float*)(ws + off); off += (size_t)BB * MM * 4;
  float* hN   = (float*)(ws + off); off += (size_t)BB * HT * 4;

  // --- weight prep (transpose + bf16) ---
  {
    int n = MM * MM;
    int g = (n + 255) / 256;
    wtrans_kernel<<<g, 256, 0, stream>>>(tq_w, tqT, MM, MM);
    wtrans_kernel<<<g, 256, 0, stream>>>(tk_w, tkT, MM, MM);
    wtrans_kernel<<<g, 256, 0, stream>>>(tv_w, tvT, MM, MM);
    wtrans_kernel<<<g, 256, 0, stream>>>(to_w, toT, MM, MM);
    int n2 = MM * 384;
    wtrans_kernel<<<(n2 + 255) / 256, 256, 0, stream>>>(gru_wih, wihT, MM, 384);
  }

  // --- stage 1+2: var embed + feature attention -> x [8192,512] f32 ---
  var_token_kernel<<<BB * SS, 128, 0, stream>>>(
      inputs, aux, mask, pre_O_w, pre_O_b, pre_M_w, pre_M_b,
      emb_O_w, emb_O_b, emb_M_w, emb_M_b, ln_v_s, ln_v_b,
      vq_w, vq_b, vk_w, vk_b, vv_w, vv_b, vo_w, vo_b, x);

  // --- temporal LN -> bf16 ---
  ln_rows_kernel<<<BB * SS, 256, 0, stream>>>(x, ln_t_s, ln_t_b, y_bf);

  // --- QKV projections (WMMA + async-LDS weight staging) ---
  dim3 gblk(32, 4);
  dim3 ggrid(BB * SS / 64, MM / 64);
  gemm_bf16_kernel<<<ggrid, gblk, 0, stream>>>(y_bf, tqT, tq_b, nullptr, 0.125f,
                                               nullptr, q_bf, BB * SS, MM, MM);
  gemm_bf16_kernel<<<ggrid, gblk, 0, stream>>>(y_bf, tkT, tk_b, nullptr, 1.0f,
                                               nullptr, k_bf, BB * SS, MM, MM);
  gemm_bf16_kernel<<<ggrid, gblk, 0, stream>>>(y_bf, tvT, tv_b, nullptr, 1.0f,
                                               vbuf, nullptr, BB * SS, MM, MM);
  {
    int n = BB * HH * 64 * SS;
    vtrans_kernel<<<(n + 255) / 256, 256, 0, stream>>>(vbuf, vt_bf);
  }

  // --- attention (WMMA, exact softmax via LDS strip) ---
  {
    size_t shmem = 16 * 1024 * 4 + 16 * 1024 * 2 + (16 + 32 + 32) * 4; // 98624 B
    flash_attn_kernel<<<BB * HH * (SS / 16), 32, shmem, stream>>>(q_bf, k_bf, vt_bf, obuf);
  }

  // --- output projection + residual -> x2 (f32 + bf16) ---
  {
    int n = BB * SS * MM;
    conv_bf16_kernel<<<(n + 255) / 256, 256, 0, stream>>>(obuf, o_bf, n);
  }
  gemm_bf16_kernel<<<ggrid, gblk, 0, stream>>>(o_bf, toT, to_b, x, 1.0f,
                                               x2, x2_bf, BB * SS, MM, MM);

  // --- GRU input projection: gx = x2 @ wih + bih ---
  {
    dim3 ggrid2(BB * SS / 64, 384 / 64);
    gemm_bf16_kernel<<<ggrid2, gblk, 0, stream>>>(x2_bf, wihT, gru_bih, nullptr, 1.0f,
                                                  gx, nullptr, BB * SS, 384, MM);
  }

  // --- GRU scan -> r_traj (into d_out) + hN ---
  {
    float* traj = out + 2 * BB * FF * DD; // after mu, logsig
    size_t shmem = (size_t)(128 * 384 + 2 * 1024) * 4; // 204800 B
    gru_scan_kernel<<<1, 1024, shmem, stream>>>(gx, gru_whh, gru_bhh, traj, hN);
  }

  // --- mean pool of r (pre-temporal x) + heads ---
  ragg_kernel<<<(BB * MM + 255) / 256, 256, 0, stream>>>(x, ragg);
  heads_kernel<<<BB * FF, DD, 0, stream>>>(ragg, hN, mu_w, mu_b, ls_w, ls_b, out);
}
